// RVQModel_59270548685251
// MI455X (gfx1250) — matrix operand order repaired and checked by reading, not compiled, not run on hardware
//
#include <hip/hip_runtime.h>

typedef __bf16 bf16;
typedef __attribute__((ext_vector_type(16))) __bf16 v16bf;
typedef __attribute__((ext_vector_type(8)))  __bf16 v8bf;
typedef __attribute__((ext_vector_type(8)))  float   v8f;

namespace {

constexpr int NTOK = 64 * 512;   // 32768 tokens
constexpr int DIN  = 150;
constexpr int DINP = 160;        // padded K for encoder-1 / N for decoder-2
constexpr int DM   = 256;
constexpr int KC   = 1024;       // codes per level
constexpr int LV   = 4;
constexpr float BETA  = 0.25f;
constexpr float QSCALE = (1.0f + BETA) / ((float)NTOK * (float)DM);

// ---- WMMA fragment loaders (bf16 16x16x32, wave32 layouts per ISA 7.12.2) ----
// A (16x32, MxK): lane = m + 16*h holds row m; elems 0..7 = K0+8h..+7, elems 8..15 = K0+16+8h..+7
__device__ inline v16bf load_a_frag(const bf16* __restrict__ row, int k0, int h) {
  v8bf lo = *reinterpret_cast<const v8bf*>(row + k0 + 8 * h);
  v8bf hi = *reinterpret_cast<const v8bf*>(row + k0 + 16 + 8 * h);
  return __builtin_shufflevector(lo, hi, 0,1,2,3,4,5,6,7,8,9,10,11,12,13,14,15);
}
// B (32x16, KxN), operand stored transposed [Ncols][K]: lane = n + 16*h holds col n,
// elems 0..7 = K0+8h..+7, elems 8..15 = K0+16+8h..+7 (same k order as A)
__device__ inline v16bf load_b_frag(const bf16* col, int k0, int h) {
  v8bf lo = *reinterpret_cast<const v8bf*>(col + k0 + 8 * h);
  v8bf hi = *reinterpret_cast<const v8bf*>(col + k0 + 16 + 8 * h);
  return __builtin_shufflevector(lo, hi, 0,1,2,3,4,5,6,7,8,9,10,11,12,13,14,15);
}

// ---------------- generic bf16 WMMA GEMM: C[M,ncols] = A[M,K] * Bt[ncols,K]^T ----------------
// block = 256 threads = 8 waves, each wave one 16x16 tile; grid = (M/128, ceil(ncolsPad/16))
__global__ __launch_bounds__(256) void gemm_bf16_wmma(
    const bf16* __restrict__ A, int lda,
    const bf16* __restrict__ Bt, int ldk,
    const float* __restrict__ bias,
    float* __restrict__ Cf, bf16* __restrict__ Cb, int ldc,
    int K, int ncols, int relu)
{
  const int lane = threadIdx.x & 31;
  const int wave = threadIdx.x >> 5;
  const int n = lane & 15;
  const int h = lane >> 4;
  const int rowBase = (blockIdx.x * 8 + wave) * 16;
  const int col = blockIdx.y * 16 + n;

  const bf16* arow = A + (size_t)(rowBase + n) * lda;  // lane's A row (m == lane&15)
  const bf16* bcol = Bt + (size_t)col * ldk;           // lane's B column

  v8f acc = {0.f, 0.f, 0.f, 0.f, 0.f, 0.f, 0.f, 0.f};
  for (int k0 = 0; k0 < K; k0 += 32) {
    v16bf a = load_a_frag(arow, k0, h);
    v16bf b = load_b_frag(bcol, k0, h);
    acc = __builtin_amdgcn_wmma_f32_16x16x32_bf16(false, a, false, b, (short)0, acc,
                                                  false, false);
  }
  if (col < ncols) {
    const float bv = bias ? bias[col] : 0.f;
    for (int v = 0; v < 8; ++v) {
      const int row = rowBase + v + 8 * h;
      float r = acc[v] + bv;
      if (relu) r = fmaxf(r, 0.f);
      if (Cf) Cf[(size_t)row * ldc + col] = r;
      if (Cb) Cb[(size_t)row * ldc + col] = (bf16)r;
    }
  }
}

// ---------------- RVQ distance + argmin for one level ----------------
// Block = 8 waves * 16 tokens. The 16-code x 256-dim codebook chunk (8 KB) is staged
// into LDS ONCE per block via async-to-LDS DMA (double buffered), then shared by all
// 8 waves -> 8x less cache traffic for the B operand of the dominant GEMM.
// dist = cnorm[c] - 2*dot(r, c)   (token ||r||^2 dropped: constant for argmin)
__global__ __launch_bounds__(256) void rvq_dist(
    const bf16* __restrict__ resb,        // [NTOK, DM] residual, bf16
    const bf16* __restrict__ cbb,         // [KC, DM] codebook (level), bf16 == Bt layout
    const float* __restrict__ cnorm,      // [KC]
    int* __restrict__ codes,              // [NTOK] out
    float* __restrict__ codes_out)        // d_out codes region for this level (stride LV)
{
  __shared__ bf16 bstage[2][16 * DM];     // 2 x 8 KB chunk buffers

  const int lane = threadIdx.x & 31;
  const int wave = threadIdx.x >> 5;
  const int n = lane & 15;
  const int h = lane >> 4;
  const int tok0 = (blockIdx.x * 8 + wave) * 16;

  // preload the full-K residual fragments (8 * v16bf = 64 VGPRs)
  v16bf afr[8];
  const bf16* arow = resb + (size_t)(tok0 + n) * DM;
  #pragma unroll
  for (int s = 0; s < 8; ++s) afr[s] = load_a_frag(arow, s * 32, h);

  float bestd[8];
  int   besti[8];
  #pragma unroll
  for (int v = 0; v < 8; ++v) { bestd[v] = 3.0e38f; besti[v] = 0; }

  // each thread DMAs 32 bytes (2 x B128) of the chunk: async global -> LDS
  const int telems = 16;                              // bf16 elems per thread
  auto issue_chunk = [&](int c, int buf) {
    unsigned ldsoff = (unsigned)(uintptr_t)(&bstage[buf][threadIdx.x * telems]);
    const bf16* gsrc = cbb + (size_t)c * (16 * DM) + threadIdx.x * telems;
    asm volatile(
        "global_load_async_to_lds_b128 %0, %1, off\n\t"
        "global_load_async_to_lds_b128 %0, %1, off offset:16"
        :: "v"(ldsoff), "v"(gsrc) : "memory");
  };

  issue_chunk(0, 0);                                  // prologue: chunk 0 in flight

  for (int c = 0; c < KC / 16; ++c) {
    const int buf = c & 1;
    // keep the pipeline full: issue next chunk (or a harmless repeat on the last one)
    issue_chunk(c + 1 < KC / 16 ? c + 1 : c, buf ^ 1);
    // 2 loads of chunk c done when <=2 remain (async loads complete in order)
    asm volatile("s_wait_asynccnt 2" ::: "memory");
    __syncthreads();                                  // chunk c visible to all waves

    const int code = c * 16 + n;
    const bf16* bcol = &bstage[buf][n * DM];
    v8f acc = {0.f, 0.f, 0.f, 0.f, 0.f, 0.f, 0.f, 0.f};
    #pragma unroll
    for (int s = 0; s < 8; ++s) {
      v16bf b = load_b_frag(bcol, s * 32, h);         // ds_load_b128 from LDS
      acc = __builtin_amdgcn_wmma_f32_16x16x32_bf16(false, afr[s], false, b, (short)0,
                                                    acc, false, false);
    }
    const float cn = cnorm[code];
    #pragma unroll
    for (int v = 0; v < 8; ++v) {
      const float d = cn - 2.0f * acc[v];
      if (d < bestd[v]) { bestd[v] = d; besti[v] = code; }
    }
    __syncthreads();                                  // done reading buf before reuse at c+2
  }
  asm volatile("s_wait_asynccnt 0" ::: "memory");     // drain tail DMA before exit

  // reduce argmin across the 16 code-lanes (stays inside each 16-lane half of the wave)
  #pragma unroll
  for (int m = 1; m < 16; m <<= 1) {
    #pragma unroll
    for (int v = 0; v < 8; ++v) {
      const float od = __shfl_xor(bestd[v], m, 32);
      const int   oi = __shfl_xor(besti[v], m, 32);
      if (od < bestd[v] || (od == bestd[v] && oi < besti[v])) {
        bestd[v] = od; besti[v] = oi;
      }
    }
  }
  if (n == 0) {
    #pragma unroll
    for (int v = 0; v < 8; ++v) {
      const int tok = tok0 + v + 8 * h;
      codes[tok] = besti[v];
      codes_out[(size_t)tok * LV] = (float)besti[v];
    }
  }
}

// ---------------- RVQ gather/update for one level (block = one token) ----------------
__global__ __launch_bounds__(256) void rvq_update(
    const float* __restrict__ cb,     // [KC, DM] codebook (level), f32
    const int* __restrict__ codes,    // [NTOK]
    float* __restrict__ res,          // [NTOK, DM] residual (in/out)
    float* __restrict__ zq,           // [NTOK, DM] accum
    bf16* __restrict__ resb,          // [NTOK, DM] bf16 residual for next level
    float* __restrict__ qloss)
{
  __shared__ float sred[256];
  const int tok = blockIdx.x;
  const int d = threadIdx.x;
  const int idx = codes[tok];
  const float q = cb[(size_t)idx * DM + d];
  const size_t o = (size_t)tok * DM + d;
  const float r = res[o] - q;
  zq[o] += q;
  res[o] = r;
  resb[o] = (bf16)r;
  sred[d] = r * r;
  __syncthreads();
  for (int s = 128; s > 0; s >>= 1) {
    if (d < s) sred[d] += sred[d + s];
    __syncthreads();
  }
  if (d == 0) atomicAdd(qloss, sred[0] * QSCALE);
}

// ---------------- prep / conversion kernels ----------------
__global__ __launch_bounds__(256) void cb_prep(const float* __restrict__ cb,
                                               bf16* __restrict__ cbb,
                                               float* __restrict__ cnorm)
{
  __shared__ float sred[256];
  const int row = blockIdx.x;            // over LV*KC rows
  const int d = threadIdx.x;
  const float v = cb[(size_t)row * DM + d];
  cbb[(size_t)row * DM + d] = (bf16)v;
  sred[d] = v * v;
  __syncthreads();
  for (int s = 128; s > 0; s >>= 1) {
    if (d < s) sred[d] += sred[d + s];
    __syncthreads();
  }
  if (d == 0) cnorm[row] = sred[0];
}

__global__ __launch_bounds__(256) void pad_x_bf16(const float* __restrict__ x,
                                                  bf16* __restrict__ xb)
{
  const int i = blockIdx.x * 256 + threadIdx.x;   // over NTOK*DINP
  if (i >= NTOK * DINP) return;
  const int t = i / DINP, k = i % DINP;
  xb[i] = (bf16)(k < DIN ? x[(size_t)t * DIN + k] : 0.f);
}

// Wt[n*Kpad + k] = W[k*N + n] (zero-padded)
__global__ __launch_bounds__(256) void transpose_pad_bf16(const float* __restrict__ W,
                                                          bf16* __restrict__ Wt,
                                                          int K, int N, int Kpad, int Npad)
{
  const int i = blockIdx.x * 256 + threadIdx.x;
  if (i >= Kpad * Npad) return;
  const int n = i / Kpad, k = i % Kpad;
  Wt[i] = (bf16)((k < K && n < N) ? W[(size_t)k * N + n] : 0.f);
}

__global__ __launch_bounds__(256) void f32_to_bf16(const float* __restrict__ s,
                                                   bf16* __restrict__ d, int n)
{
  const int i = blockIdx.x * 256 + threadIdx.x;
  if (i < n) d[i] = (bf16)s[i];
}

inline size_t alignup(size_t x) { return (x + 255) & ~(size_t)255; }

} // namespace

extern "C" void kernel_launch(void* const* d_in, const int* in_sizes, int n_in,
                              void* d_out, int out_size, void* d_ws, size_t ws_size,
                              hipStream_t stream) {
  (void)in_sizes; (void)n_in; (void)out_size; (void)ws_size;
  const float* x      = (const float*)d_in[0];
  const float* enc_w1 = (const float*)d_in[1];
  const float* enc_b1 = (const float*)d_in[2];
  const float* enc_w2 = (const float*)d_in[3];
  const float* enc_b2 = (const float*)d_in[4];
  const float* dec_w1 = (const float*)d_in[5];
  const float* dec_b1 = (const float*)d_in[6];
  const float* dec_w2 = (const float*)d_in[7];
  const float* dec_b2 = (const float*)d_in[8];
  const float* cbooks = (const float*)d_in[9];

  float* out = (float*)d_out;
  float* out_recon = out;                              // [NTOK, DIN]
  float* out_codes = out + (size_t)NTOK * DIN;         // [NTOK, LV] (as f32)
  float* out_qloss = out_codes + (size_t)NTOK * LV;    // scalar

  // -------- workspace carve-out --------
  char* p = (char*)d_ws;
  size_t off = 0;
  auto take = [&](size_t bytes) { char* r = p + off; off = alignup(off + bytes); return r; };
  bf16*  xb    = (bf16*) take((size_t)NTOK * DINP * 2);
  bf16*  w1t   = (bf16*) take((size_t)DM * DINP * 2);     // [256][160]
  bf16*  w2t   = (bf16*) take((size_t)DM * DM * 2);
  bf16*  dw1t  = (bf16*) take((size_t)DM * DM * 2);
  bf16*  dw2t  = (bf16*) take((size_t)DINP * DM * 2);     // [160][256]
  bf16*  hbuf  = (bf16*) take((size_t)NTOK * DM * 2);     // enc hidden; reused as dec hidden g
  float* res   = (float*)take((size_t)NTOK * DM * 4);     // residual (starts as z)
  bf16*  resb  = (bf16*) take((size_t)NTOK * DM * 2);
  float* zq    = (float*)take((size_t)NTOK * DM * 4);
  bf16*  zqb   = (bf16*) take((size_t)NTOK * DM * 2);
  bf16*  cbb   = (bf16*) take((size_t)LV * KC * DM * 2);
  float* cnorm = (float*)take((size_t)LV * KC * 4);
  int*   codes = (int*)  take((size_t)NTOK * 4);

  // -------- init --------
  hipMemsetAsync(zq, 0, (size_t)NTOK * DM * 4, stream);
  hipMemsetAsync(out_qloss, 0, 4, stream);

  // -------- operand prep --------
  cb_prep<<<LV * KC, 256, 0, stream>>>(cbooks, cbb, cnorm);
  pad_x_bf16<<<(NTOK * DINP + 255) / 256, 256, 0, stream>>>(x, xb);
  transpose_pad_bf16<<<(DM * DINP + 255) / 256, 256, 0, stream>>>(enc_w1, w1t, DIN, DM, DINP, DM);
  transpose_pad_bf16<<<(DM * DM + 255) / 256, 256, 0, stream>>>(enc_w2, w2t, DM, DM, DM, DM);
  transpose_pad_bf16<<<(DM * DM + 255) / 256, 256, 0, stream>>>(dec_w1, dw1t, DM, DM, DM, DM);
  transpose_pad_bf16<<<(DINP * DM + 255) / 256, 256, 0, stream>>>(dec_w2, dw2t, DM, DIN, DM, DINP);

  const dim3 blk(256);
  const int mBlocks = NTOK / 128;   // 8 waves * 16 rows per block

  // -------- encoder --------
  gemm_bf16_wmma<<<dim3(mBlocks, DM / 16), blk, 0, stream>>>(
      xb, DINP, w1t, DINP, enc_b1, nullptr, hbuf, DM, DINP, DM, 1);
  gemm_bf16_wmma<<<dim3(mBlocks, DM / 16), blk, 0, stream>>>(
      hbuf, DM, w2t, DM, enc_b2, res, nullptr, DM, DM, DM, 0);
  f32_to_bf16<<<(NTOK * DM + 255) / 256, blk, 0, stream>>>(res, resb, NTOK * DM);

  // -------- RVQ levels --------
  for (int l = 0; l < LV; ++l) {
    rvq_dist<<<NTOK / 128, blk, 0, stream>>>(
        resb, cbb + (size_t)l * KC * DM, cnorm + (size_t)l * KC, codes, out_codes + l);
    rvq_update<<<NTOK, blk, 0, stream>>>(
        cbooks + (size_t)l * KC * DM, codes, res, zq, resb, out_qloss);
  }
  f32_to_bf16<<<(NTOK * DM + 255) / 256, blk, 0, stream>>>(zq, zqb, NTOK * DM);

  // -------- decoder (straight-through forward value == z_q) --------
  gemm_bf16_wmma<<<dim3(mBlocks, DM / 16), blk, 0, stream>>>(
      zqb, DM, dw1t, DM, dec_b1, nullptr, hbuf, DM, DM, DM, 1);
  gemm_bf16_wmma<<<dim3(mBlocks, DINP / 16), blk, 0, stream>>>(
      hbuf, DM, dw2t, DM, dec_b2, out_recon, nullptr, DIN, DM, DIN, 0);
}
